// Qwen3NextAttention_3985729651555
// MI455X (gfx1250) — matrix operand order repaired and checked by reading, not compiled, not run on hardware
//
#include <hip/hip_runtime.h>

// ---------------- problem constants ----------------
constexpr int B_  = 2;
constexpr int S_  = 2048;
constexpr int D_  = 2048;
constexpr int H_  = 16;
constexpr int KV_ = 4;
constexpr int HD_ = 128;
constexpr int NT  = B_ * S_;        // 4096 tokens
constexpr int QO  = H_ * HD_ * 2;   // 4096 (q + gate)
constexpr int KO  = KV_ * HD_;      // 512

typedef __attribute__((ext_vector_type(16))) __bf16 bf16x16;
typedef __attribute__((ext_vector_type(8)))  float  v8f;

union Frag { bf16x16 v; uint4 u[2]; };

static __device__ inline v8f vzero() {
  v8f v = {0.f, 0.f, 0.f, 0.f, 0.f, 0.f, 0.f, 0.f};
  return v;
}

// ---- CDNA5 async global->LDS copy (32 bytes: two B128 transfers) ----
// INST_OFFSET is added to BOTH the LDS and global addresses, so one LDS-addr
// VGPR + offset:16 covers a contiguous 32B chunk.
static __device__ inline void async_copy32(unsigned lds, const __bf16* g) {
  unsigned long long ga = (unsigned long long)(uintptr_t)g;
  asm volatile("global_load_async_to_lds_b128 %0, %1, off\n\t"
               "global_load_async_to_lds_b128 %0, %1, off offset:16"
               :
               : "v"(lds), "v"(ga)
               : "memory");
}
static __device__ inline void async_waitall() {
  asm volatile("s_wait_asynccnt 0x0" ::: "memory");
}

// ---------------- f32 -> bf16 convert ----------------
__global__ void cvt_kernel(const float* __restrict__ in, __bf16* __restrict__ out, int n) {
  int i = blockIdx.x * blockDim.x + threadIdx.x;
  if (i < n) out[i] = (__bf16)in[i];
}

// ---------------- bf16 GEMM: C[M,N] = A[M,K] * W[N,K]^T ----------------
// 128x128 block tile, 8 waves (4x2); each wave computes 32x64 (2x4 WMMA tiles).
// K stepped by 32 through double-buffered LDS filled with async copies.
__global__ __launch_bounds__(256) void gemm_bf16_nt(const __bf16* __restrict__ A,
                                                    const __bf16* __restrict__ W,
                                                    float* __restrict__ C,
                                                    int M, int N, int K) {
  __shared__ __bf16 As[2][128][32];
  __shared__ __bf16 Ws[2][128][32];

  const int t    = threadIdx.x;
  const int m0   = blockIdx.y * 128;
  const int n0   = blockIdx.x * 128;
  const int wv   = t >> 5;
  const int lane = t & 31;
  const int ln   = lane & 15;        // tile column (N) / row index
  const int hs   = lane >> 4;        // half-wave select
  const int wm   = (wv >> 1) * 32;   // wave M offset within block tile
  const int wn   = (wv & 1) * 64;    // wave N offset within block tile

  // loader mapping: each thread owns one 32B chunk of the 128x32 slab
  const int lrow  = t >> 1;          // 0..127
  const int lcol  = (t & 1) * 16;    // 0 or 16

  const unsigned asBase = (unsigned)(uintptr_t)&As[0][0][0];
  const unsigned wsBase = (unsigned)(uintptr_t)&Ws[0][0][0];
  const unsigned slabBytes = 128 * 32 * 2;   // 8KB per buffer
  const unsigned ldsOff = (unsigned)(lrow * 32 + lcol) * 2;

  const __bf16* gA = A + (size_t)(m0 + lrow) * K + lcol;
  const __bf16* gW = W + (size_t)(n0 + lrow) * K + lcol;

  v8f acc[2][4];
#pragma unroll
  for (int i = 0; i < 2; ++i)
#pragma unroll
    for (int j = 0; j < 4; ++j) acc[i][j] = vzero();

  // prologue: fill buffer 0
  async_copy32(asBase + ldsOff, gA);
  async_copy32(wsBase + ldsOff, gW);

  int cur = 0;
  for (int kb = 0; kb < K; kb += 32) {
    async_waitall();        // my slab writes (issued last iteration) are in LDS
    __syncthreads();        // everyone's writes visible; prior reads retired

    if (kb + 32 < K) {
      const unsigned nb = (cur ^ 1) * slabBytes;
      async_copy32(asBase + nb + ldsOff, gA + kb + 32);
      async_copy32(wsBase + nb + ldsOff, gW + kb + 32);
    }

    // A fragments: row = wm + i*16 + ln, K-chunks {8hs..8hs+7, 16+8hs..}
    Frag af[2];
#pragma unroll
    for (int i = 0; i < 2; ++i) {
      const __bf16* ap = &As[cur][wm + i * 16 + ln][hs * 8];
      af[i].u[0] = *reinterpret_cast<const uint4*>(ap);
      af[i].u[1] = *reinterpret_cast<const uint4*>(ap + 16);
    }
    // B fragments: row n, K-chunks {8hs..8hs+7, 16+8hs..}
    Frag bf[4];
#pragma unroll
    for (int j = 0; j < 4; ++j) {
      const __bf16* wp = &Ws[cur][wn + j * 16 + ln][hs * 8];
      bf[j].u[0] = *reinterpret_cast<const uint4*>(wp);
      bf[j].u[1] = *reinterpret_cast<const uint4*>(wp + 16);
    }
#pragma unroll
    for (int i = 0; i < 2; ++i)
#pragma unroll
      for (int j = 0; j < 4; ++j)
        acc[i][j] = __builtin_amdgcn_wmma_f32_16x16x32_bf16(
            false, af[i].v, false, bf[j].v, (short)0, acc[i][j], false, false);

    __syncthreads();        // my reads of 'cur' done before it is refilled
    cur ^= 1;
  }

  // C layout: VGPR r -> row r (lanes 0-15) / 8+r (lanes 16-31), col = ln
#pragma unroll
  for (int i = 0; i < 2; ++i) {
    const int row = m0 + wm + i * 16 + hs * 8;
#pragma unroll
    for (int j = 0; j < 4; ++j) {
      const int col = n0 + wn + j * 16 + ln;
#pragma unroll
      for (int r = 0; r < 8; ++r) {
        C[(size_t)(row + r) * N + col] = acc[i][j][r];
      }
    }
  }
}

// ---------------- RMSNorm + RoPE (q and k), writes bf16 [B, NH, S, HD] ----------------
__global__ __launch_bounds__(128) void rmsnorm_rope_kernel(const float* __restrict__ in,
                                                           const float* __restrict__ nw,
                                                           const float* __restrict__ freqs,
                                                           __bf16* __restrict__ out,
                                                           int nheads, int rowStride,
                                                           int headStride, float outScale) {
  const int i = threadIdx.x;                 // 0..127
  const int s = blockIdx.x, h = blockIdx.y, b = blockIdx.z;
  const float* src = in + (size_t)(b * S_ + s) * rowStride + h * headStride;

  float x  = src[i];
  float ss = x * x;
#pragma unroll
  for (int m = 16; m >= 1; m >>= 1) ss += __shfl_xor(ss, m, 32);

  __shared__ float red[4];
  __shared__ float nrm[128];
  if ((i & 31) == 0) red[i >> 5] = ss;
  __syncthreads();
  float tot   = red[0] + red[1] + red[2] + red[3];
  float scale = rsqrtf(tot * (1.0f / 128.0f) + 1e-6f);
  nrm[i] = x * scale * (1.0f + nw[i]);
  __syncthreads();

  const int j = i & 63;
  float c  = freqs[(size_t)s * 128 + j];
  float sn = freqs[(size_t)s * 128 + 64 + j];
  float o  = (i < 64) ? (nrm[i] * c - nrm[i + 64] * sn)
                      : (nrm[i] * c + nrm[i - 64] * sn);
  out[((size_t)(b * nheads + h) * S_ + s) * HD_ + i] = (__bf16)(o * outScale);
}

// ---------------- V transpose: vproj f32 [tok, KV*HD] -> vt bf16 [B, KV, HD, S] ----------------
__global__ void vtranspose_kernel(const float* __restrict__ vproj, __bf16* __restrict__ vt) {
  size_t idx   = (size_t)blockIdx.x * blockDim.x + threadIdx.x;
  size_t total = (size_t)B_ * KV_ * HD_ * S_;
  if (idx >= total) return;
  int s  = (int)(idx % S_);
  int f  = (int)((idx / S_) % HD_);
  int kv = (int)((idx / ((size_t)S_ * HD_)) % KV_);
  int b  = (int)(idx / ((size_t)S_ * HD_ * KV_));
  vt[idx] = (__bf16)vproj[(size_t)(b * S_ + s) * (KV_ * HD_) + kv * HD_ + f];
}

// ---------------- causal flash attention: one wave per (b, h, 16-row q tile) ----------------
__global__ __launch_bounds__(32) void flash_kernel(const __bf16* __restrict__ qb,
                                                   const __bf16* __restrict__ kbuf,
                                                   const __bf16* __restrict__ vtb,
                                                   float* __restrict__ attn) {
  const int lane  = threadIdx.x;
  const int qt    = blockIdx.x, h = blockIdx.y, b = blockIdx.z;
  const int qbase = qt * 16;
  const int kv    = h >> 2;          // GQA group = 4
  const int ln    = lane & 15;
  const int hs    = lane >> 4;

  const __bf16* qp = qb  + (size_t)(b * H_ + h)  * S_ * HD_;
  const __bf16* kp = kbuf + (size_t)(b * KV_ + kv) * S_ * HD_;
  const __bf16* vp = vtb + (size_t)(b * KV_ + kv) * HD_ * S_;

  // Q fragments (A layout), 4 chunks of K=32 covering HD=128; scale pre-baked.
  Frag qf[4];
  {
    const __bf16* base = qp + (size_t)(qbase + ln) * HD_;
    const int kb_l = hs * 8;
#pragma unroll
    for (int c = 0; c < 4; ++c) {
      qf[c].u[0] = *reinterpret_cast<const uint4*>(base + c * 32 + kb_l);
      qf[c].u[1] = *reinterpret_cast<const uint4*>(base + c * 32 + 16 + kb_l);
    }
  }

  v8f O[8];
#pragma unroll
  for (int f = 0; f < 8; ++f) O[f] = vzero();
  float mrow[8], lrow[8];
#pragma unroll
  for (int r = 0; r < 8; ++r) { mrow[r] = -1e30f; lrow[r] = 0.0f; }

  __shared__ __bf16 p_lds[16][32];

  for (int j0 = 0; j0 < qbase + 16; j0 += 32) {
    // ---- scores: two 16x16 tiles over a 32-key chunk, K=128 in 4 WMMAs each
    v8f Sc[2];
    Sc[0] = vzero(); Sc[1] = vzero();
#pragma unroll
    for (int n = 0; n < 2; ++n) {
#pragma unroll
      for (int c = 0; c < 4; ++c) {
        Frag kf;
        const __bf16* kbp = kp + (size_t)(j0 + n * 16 + ln) * HD_ + c * 32 + hs * 8;
        kf.u[0] = *reinterpret_cast<const uint4*>(kbp);
        kf.u[1] = *reinterpret_cast<const uint4*>(kbp + 16);
        Sc[n] = __builtin_amdgcn_wmma_f32_16x16x32_bf16(false, qf[c].v, false, kf.v,
                                                        (short)0, Sc[n], false, false);
      }
    }

    // ---- causal mask + online softmax (rows live in 16-lane halves)
    float mnew[8];
#pragma unroll
    for (int r = 0; r < 8; ++r) {
      const int qrow = qbase + r + hs * 8;
      float s0 = Sc[0][r];
      float s1 = Sc[1][r];
      if (j0 + ln > qrow)      s0 = -1e30f;
      if (j0 + 16 + ln > qrow) s1 = -1e30f;
      Sc[0][r] = s0; Sc[1][r] = s1;
      float mx = fmaxf(s0, s1);
#pragma unroll
      for (int m = 8; m >= 1; m >>= 1) mx = fmaxf(mx, __shfl_xor(mx, m, 32));
      mnew[r] = fmaxf(mrow[r], mx);
    }

#pragma unroll
    for (int r = 0; r < 8; ++r) {
      float alpha = __expf(mrow[r] - mnew[r]);
      float p0 = __expf(Sc[0][r] - mnew[r]);
      float p1 = __expf(Sc[1][r] - mnew[r]);
      float rs = p0 + p1;
#pragma unroll
      for (int m = 8; m >= 1; m >>= 1) rs += __shfl_xor(rs, m, 32);
      lrow[r] = lrow[r] * alpha + rs;
      mrow[r] = mnew[r];
#pragma unroll
      for (int f = 0; f < 8; ++f) O[f][r] *= alpha;
      p_lds[r + hs * 8][ln]      = (__bf16)p0;
      p_lds[r + hs * 8][16 + ln] = (__bf16)p1;
    }
    __syncthreads();

    // ---- P (16x32) A-fragment from LDS
    Frag pf;
    {
      const __bf16* pr = &p_lds[ln][hs * 8];
      pf.u[0] = *reinterpret_cast<const uint4*>(pr);
      pf.u[1] = *reinterpret_cast<const uint4*>(pr + 16);
    }
    __syncthreads();

    // ---- O += P @ V  (8 feature tiles of 16)
#pragma unroll
    for (int f = 0; f < 8; ++f) {
      Frag vf;
      const __bf16* vbp = vp + (size_t)(f * 16 + ln) * S_ + j0 + hs * 8;
      vf.u[0] = *reinterpret_cast<const uint4*>(vbp);
      vf.u[1] = *reinterpret_cast<const uint4*>(vbp + 16);
      O[f] = __builtin_amdgcn_wmma_f32_16x16x32_bf16(false, pf.v, false, vf.v,
                                                     (short)0, O[f], false, false);
    }
  }

  float inv[8];
#pragma unroll
  for (int r = 0; r < 8; ++r) inv[r] = 1.0f / lrow[r];
#pragma unroll
  for (int f = 0; f < 8; ++f) {
#pragma unroll
    for (int r = 0; r < 8; ++r) {
      attn[(size_t)(b * S_ + qbase + r + hs * 8) * (H_ * HD_) + h * HD_ + f * 16 + ln] =
          O[f][r] * inv[r];
    }
  }
}

// ---------------- gate: attn_bf16 = bf16(attn * sigmoid(gate)) ----------------
__global__ void gate_kernel(const float* __restrict__ attnf, const float* __restrict__ qg,
                            __bf16* __restrict__ out) {
  size_t idx   = (size_t)blockIdx.x * blockDim.x + threadIdx.x;
  size_t total = (size_t)NT * (H_ * HD_);
  if (idx >= total) return;
  int tok = (int)(idx / (H_ * HD_));
  int i   = (int)(idx % (H_ * HD_));
  float g   = qg[(size_t)tok * QO + (i >> 7) * 256 + 128 + (i & 127)];
  float sig = 1.0f / (1.0f + __expf(-g));
  out[idx] = (__bf16)(attnf[idx] * sig);
}

// ---------------- host launch ----------------
extern "C" void kernel_launch(void* const* d_in, const int* in_sizes, int n_in,
                              void* d_out, int out_size, void* d_ws, size_t ws_size,
                              hipStream_t stream) {
  const float* x     = (const float*)d_in[0];
  const float* freqs = (const float*)d_in[1];
  const float* Wq    = (const float*)d_in[2];
  const float* Wk    = (const float*)d_in[3];
  const float* Wv    = (const float*)d_in[4];
  const float* Wo    = (const float*)d_in[5];
  const float* qnw   = (const float*)d_in[6];
  const float* knw   = (const float*)d_in[7];
  float* out = (float*)d_out;

  char* w = (char*)d_ws;
  auto alloc = [&](size_t bytes) -> char* {
    char* p = w;
    w += (bytes + 255) & ~(size_t)255;
    return p;
  };
  __bf16* x_b    = (__bf16*)alloc((size_t)NT * D_ * 2);
  __bf16* wq_b   = (__bf16*)alloc((size_t)QO * D_ * 2);
  __bf16* wk_b   = (__bf16*)alloc((size_t)KO * D_ * 2);
  __bf16* wv_b   = (__bf16*)alloc((size_t)KO * D_ * 2);
  __bf16* wo_b   = (__bf16*)alloc((size_t)D_ * (H_ * HD_) * 2);
  float*  qg     = (float*)alloc((size_t)NT * QO * 4);
  float*  kproj  = (float*)alloc((size_t)NT * KO * 4);
  float*  vproj  = (float*)alloc((size_t)NT * KO * 4);
  __bf16* q_b    = (__bf16*)alloc((size_t)B_ * H_ * S_ * HD_ * 2);
  __bf16* k_b    = (__bf16*)alloc((size_t)B_ * KV_ * S_ * HD_ * 2);
  __bf16* vt_b   = (__bf16*)alloc((size_t)B_ * KV_ * HD_ * S_ * 2);
  float*  attnf  = (float*)alloc((size_t)NT * (H_ * HD_) * 4);
  __bf16* attn_b = (__bf16*)alloc((size_t)NT * (H_ * HD_) * 2);

  auto cvt = [&](const float* in, __bf16* o, size_t n) {
    cvt_kernel<<<(unsigned)((n + 255) / 256), 256, 0, stream>>>(in, o, (int)n);
  };
  cvt(x,  x_b,  (size_t)NT * D_);
  cvt(Wq, wq_b, (size_t)QO * D_);
  cvt(Wk, wk_b, (size_t)KO * D_);
  cvt(Wv, wv_b, (size_t)KO * D_);
  cvt(Wo, wo_b, (size_t)D_ * (H_ * HD_));

  // projections (block tile 128x128)
  gemm_bf16_nt<<<dim3(QO / 128, NT / 128), 256, 0, stream>>>(x_b, wq_b, qg,    NT, QO, D_);
  gemm_bf16_nt<<<dim3(KO / 128, NT / 128), 256, 0, stream>>>(x_b, wk_b, kproj, NT, KO, D_);
  gemm_bf16_nt<<<dim3(KO / 128, NT / 128), 256, 0, stream>>>(x_b, wv_b, vproj, NT, KO, D_);

  // norm + rope (q gets 1/sqrt(HD) baked in)
  rmsnorm_rope_kernel<<<dim3(S_, H_, B_), 128, 0, stream>>>(
      qg, qnw, freqs, q_b, H_, QO, 2 * HD_, 0.08838834764831845f);
  rmsnorm_rope_kernel<<<dim3(S_, KV_, B_), 128, 0, stream>>>(
      kproj, knw, freqs, k_b, KV_, KO, HD_, 1.0f);
  vtranspose_kernel<<<(B_ * KV_ * HD_ * S_ + 255) / 256, 256, 0, stream>>>(vproj, vt_b);

  // causal flash attention
  flash_kernel<<<dim3(S_ / 16, H_, B_), 32, 0, stream>>>(q_b, k_b, vt_b, attnf);

  // gate + convert
  gate_kernel<<<(unsigned)(((size_t)NT * H_ * HD_ + 255) / 256), 256, 0, stream>>>(
      attnf, qg, attn_b);

  // output projection -> d_out
  gemm_bf16_nt<<<dim3(D_ / 128, NT / 128), 256, 0, stream>>>(attn_b, wo_b, out, NT, D_,
                                                             H_ * HD_);
}
